// TopKSAE_11656541241426
// MI455X (gfx1250) — matrix operand order repaired and checked by reading, not compiled, not run on hardware
//
#include <hip/hip_runtime.h>

// Problem sizes (fixed by the reference)
#define NB    16384
#define DIN   768
#define DHID  12288
#define TOPK_ 64

typedef __attribute__((ext_vector_type(16))) __bf16 v16bf;
typedef __attribute__((ext_vector_type(8)))  float  v8f;
typedef __attribute__((ext_vector_type(4)))  int    v4i;
typedef __attribute__((address_space(1))) v4i* gp_v4i;   // global (AS1) v4i*
typedef __attribute__((address_space(3))) v4i* lp_v4i;   // LDS (AS3) v4i*

union FragAB { uint4 u[2]; v16bf v; };

#if defined(__has_builtin)
#if __has_builtin(__builtin_amdgcn_global_load_async_to_lds_b128)
#define HAVE_ASYNC_LDS 1
#endif
#endif
#ifndef HAVE_ASYNC_LDS
#define HAVE_ASYNC_LDS 0
#endif

__device__ __forceinline__ unsigned short f2bf_rn(float f) {
  unsigned u = __float_as_uint(f);
  u += 0x7FFFu + ((u >> 16) & 1u);        // round-to-nearest-even
  return (unsigned short)(u >> 16);
}
__device__ __forceinline__ float bf2f(unsigned short h) {
  return __uint_as_float(((unsigned)h) << 16);
}
__device__ __forceinline__ void split2(float v, unsigned short& hi, unsigned short& lo) {
  hi = f2bf_rn(v);
  lo = f2bf_rn(v - bf2f(hi));
}

// 16-byte global -> LDS copy: async on gfx1250 if the builtin exists.
// AS casts done via integers: generic global == AS1 numerically; a generic LDS
// pointer's low 32 bits are the AS3 offset (ISA 10.2).
__device__ __forceinline__ void cp16(const unsigned short* g, unsigned short* l) {
#if HAVE_ASYNC_LDS
  gp_v4i gp = (gp_v4i)(unsigned long long)g;
  lp_v4i lp = (lp_v4i)(unsigned int)(unsigned long long)l;
  __builtin_amdgcn_global_load_async_to_lds_b128(gp, lp, 0, 0);
#else
  *(uint4*)l = *(const uint4*)g;
#endif
}
__device__ __forceinline__ void async_wait_prev16() {
#if HAVE_ASYNC_LDS
#if defined(__has_builtin) && __has_builtin(__builtin_amdgcn_s_wait_asynccnt)
  __builtin_amdgcn_s_wait_asynccnt(16);
#else
  asm volatile("s_wait_asynccnt 0x10" ::: "memory");
#endif
#endif
}
__device__ __forceinline__ void async_wait_all() {
#if HAVE_ASYNC_LDS
#if defined(__has_builtin) && __has_builtin(__builtin_amdgcn_s_wait_asynccnt)
  __builtin_amdgcn_s_wait_asynccnt(0);
#else
  asm volatile("s_wait_asynccnt 0x0" ::: "memory");
#endif
#endif
}

// ---------------------------------------------------------------------------
// Kernel S: split fp32 matrix into hi/lo bf16 planes (one-time, removes all
// conversion VALU from the GEMM hot loop).
// ---------------------------------------------------------------------------
__global__ __launch_bounds__(256) void k_split(const float* __restrict__ src,
                                               unsigned short* __restrict__ hi,
                                               unsigned short* __restrict__ lo,
                                               int n4) {
  const int i = blockIdx.x * 256 + threadIdx.x;
  if (i >= n4) return;
  const float4 v = ((const float4*)src)[i];
  unsigned short h0, h1, h2, h3, l0, l1, l2, l3;
  split2(v.x, h0, l0); split2(v.y, h1, l1);
  split2(v.z, h2, l2); split2(v.w, h3, l3);
  uint2 ph, pl;
  ph.x = (unsigned)h0 | ((unsigned)h1 << 16); ph.y = (unsigned)h2 | ((unsigned)h3 << 16);
  pl.x = (unsigned)l0 | ((unsigned)l1 << 16); pl.y = (unsigned)l2 | ((unsigned)l3 << 16);
  ((uint2*)hi)[i] = ph;
  ((uint2*)lo)[i] = pl;
}

// ---------------------------------------------------------------------------
// Kernel 0: transpose W_dec [DIN][DHID] -> WT [DHID][DIN] (for coalesced decode)
// ---------------------------------------------------------------------------
__global__ __launch_bounds__(256) void k_transpose_wdec(const float* __restrict__ W,
                                                        float* __restrict__ WT) {
  __shared__ float tile[32][33];
  const int tx = threadIdx.x & 31;
  const int ty = threadIdx.x >> 5;
  const int h0 = blockIdx.x * 32;
  const int d0 = blockIdx.y * 32;
#pragma unroll
  for (int j = 0; j < 4; ++j)
    tile[ty + j * 8][tx] = W[(size_t)(d0 + ty + j * 8) * DHID + h0 + tx];
  __syncthreads();
#pragma unroll
  for (int j = 0; j < 4; ++j)
    WT[(size_t)(h0 + ty + j * 8) * DIN + d0 + tx] = tile[tx][ty + j * 8];
}

// ---------------------------------------------------------------------------
// Kernel 1a (preferred): encode GEMM from pre-split bf16 planes.
//   z = relu(x @ W_enc^T + b_enc), split-bf16 (hi*hi + hi*lo + lo*hi) in
//   v_wmma_f32_16x16x32_bf16 with fp32 accumulators.
//   Block tile 128x128, K-tile 64, double-buffered LDS (128 KB, dynamic),
//   tiles fetched with GLOBAL_LOAD_ASYNC_TO_LDS_B128 (ASYNCcnt pipelined).
//   LDS plane layout (ushort units): AsH@0, AsL@16384, BsH@32768, BsL@49152;
//   each plane has two 8192-element buffers (buf*8192).
// ---------------------------------------------------------------------------
__global__ __launch_bounds__(256) void k_encode_pre(const unsigned short* __restrict__ xH,
                                                    const unsigned short* __restrict__ xL,
                                                    const unsigned short* __restrict__ WeH,
                                                    const unsigned short* __restrict__ WeL,
                                                    const float* __restrict__ be,
                                                    float* __restrict__ z) {
  extern __shared__ unsigned short smem[];       // 8 planes x 8192 ushort = 128 KB

  const int tid  = threadIdx.x;
  const int lane = tid & 31;
  const int wave = tid >> 5;
  const int wm   = wave & 1;
  const int wn   = wave >> 1;
  const int rowBase = blockIdx.y * 128;
  const int colBase = blockIdx.x * 128;

  auto issue_tile = [&](int buf, int k0) {
    unsigned short* aH = smem + 0 * 16384 + buf * 8192;
    unsigned short* aL = smem + 1 * 16384 + buf * 8192;
    unsigned short* bH = smem + 2 * 16384 + buf * 8192;
    unsigned short* bL = smem + 3 * 16384 + buf * 8192;
#pragma unroll
    for (int i = 0; i < 4; ++i) {
      const int id = tid + i * 256;              // 1024 chunks of 8 bf16
      const int r  = id >> 3;                    // 0..127
      const int c  = (id & 7) << 3;              // 0..56 step 8
      const size_t ga = (size_t)(rowBase + r) * DIN + k0 + c;
      const size_t gb = (size_t)(colBase + r) * DIN + k0 + c;
      const int loff = r * 64 + c;
      cp16(xH  + ga, aH + loff);
      cp16(xL  + ga, aL + loff);
      cp16(WeH + gb, bH + loff);
      cp16(WeL + gb, bL + loff);
    }
  };

  v8f acc[4][2];
#pragma unroll
  for (int mi = 0; mi < 4; ++mi)
#pragma unroll
    for (int ni = 0; ni < 2; ++ni) acc[mi][ni] = {};

  issue_tile(0, 0);

  const int NKT = DIN / 64;                      // 12
  for (int kt = 0; kt < NKT; ++kt) {
    const int cur = kt & 1;
    const int k0n = (kt + 1) * 64;
    const bool more = (k0n < DIN);
    if (more) {
      issue_tile(cur ^ 1, k0n);
      async_wait_prev16();                       // this wave's cur-tile ops done
    } else {
      async_wait_all();
    }
    __syncthreads();                             // everyone's cur tile in LDS

    const unsigned short* aHc = smem + 0 * 16384 + cur * 8192;
    const unsigned short* aLc = smem + 1 * 16384 + cur * 8192;
    const unsigned short* bHc = smem + 2 * 16384 + cur * 8192;
    const unsigned short* bLc = smem + 3 * 16384 + cur * 8192;

#pragma unroll
    for (int ks = 0; ks < 64; ks += 32) {
      FragAB ah[4], al[4], bh[2], bl[2];
      const int am  = lane & 15;
      const int akb = ks + ((lane >> 4) << 3);
#pragma unroll
      for (int mi = 0; mi < 4; ++mi) {
        const int r = (wm * 64 + mi * 16 + am) * 64;
        ah[mi].u[0] = *(const uint4*)(aHc + r + akb);
        ah[mi].u[1] = *(const uint4*)(aHc + r + akb + 16);
        al[mi].u[0] = *(const uint4*)(aLc + r + akb);
        al[mi].u[1] = *(const uint4*)(aLc + r + akb + 16);
      }
      const int bn  = lane & 15;
      const int bkb = ks + ((lane >> 4) << 3);
#pragma unroll
      for (int ni = 0; ni < 2; ++ni) {
        const int cc = (wn * 32 + ni * 16 + bn) * 64;
        bh[ni].u[0] = *(const uint4*)(bHc + cc + bkb);
        bh[ni].u[1] = *(const uint4*)(bHc + cc + bkb + 16);
        bl[ni].u[0] = *(const uint4*)(bLc + cc + bkb);
        bl[ni].u[1] = *(const uint4*)(bLc + cc + bkb + 16);
      }
#pragma unroll
      for (int mi = 0; mi < 4; ++mi)
#pragma unroll
        for (int ni = 0; ni < 2; ++ni) {
          acc[mi][ni] = __builtin_amdgcn_wmma_f32_16x16x32_bf16(
              false, ah[mi].v, false, bh[ni].v, (short)0, acc[mi][ni], false, false);
          acc[mi][ni] = __builtin_amdgcn_wmma_f32_16x16x32_bf16(
              false, ah[mi].v, false, bl[ni].v, (short)0, acc[mi][ni], false, false);
          acc[mi][ni] = __builtin_amdgcn_wmma_f32_16x16x32_bf16(
              false, al[mi].v, false, bh[ni].v, (short)0, acc[mi][ni], false, false);
        }
    }
    __syncthreads();                             // done reading cur before overwrite
  }

#pragma unroll
  for (int ni = 0; ni < 2; ++ni) {
    const int n = colBase + wn * 32 + ni * 16 + (lane & 15);
    const float bias = be[n];
#pragma unroll
    for (int mi = 0; mi < 4; ++mi) {
      const int mb = rowBase + wm * 64 + mi * 16 + ((lane >> 4) << 3);
#pragma unroll
      for (int r = 0; r < 8; ++r) {
        float v = acc[mi][ni][r] + bias;
        v = v > 0.0f ? v : 0.0f;
        z[(size_t)(mb + r) * DHID + n] = v;
      }
    }
  }
}

// ---------------------------------------------------------------------------
// Kernel 1b (fallback, small ws): encode GEMM with in-kernel fp32->bf16 split.
// ---------------------------------------------------------------------------
__global__ __launch_bounds__(256) void k_encode(const float* __restrict__ x,
                                                const float* __restrict__ We,
                                                const float* __restrict__ be,
                                                float* __restrict__ z) {
  __shared__ unsigned short AsH[128][64];
  __shared__ unsigned short AsL[128][64];
  __shared__ unsigned short BsH[128][64];
  __shared__ unsigned short BsL[128][64];

  const int tid  = threadIdx.x;
  const int lane = tid & 31;
  const int wave = tid >> 5;
  const int wm   = wave & 1;
  const int wn   = wave >> 1;
  const int rowBase = blockIdx.y * 128;
  const int colBase = blockIdx.x * 128;

  v8f acc[4][2];
#pragma unroll
  for (int mi = 0; mi < 4; ++mi)
#pragma unroll
    for (int ni = 0; ni < 2; ++ni) acc[mi][ni] = {};

  for (int k0 = 0; k0 < DIN; k0 += 64) {
#pragma unroll
    for (int i = 0; i < 8; ++i) {
      const int id = tid + i * 256;
      const int r  = id >> 4;
      const int c  = (id & 15) << 2;
      const float4 av = *(const float4*)(x  + (size_t)(rowBase + r) * DIN + k0 + c);
      const float4 bv = *(const float4*)(We + (size_t)(colBase + r) * DIN + k0 + c);
      unsigned short h0, h1, h2, h3, l0, l1, l2, l3;
      split2(av.x, h0, l0); split2(av.y, h1, l1);
      split2(av.z, h2, l2); split2(av.w, h3, l3);
      uint2 ph, pl;
      ph.x = (unsigned)h0 | ((unsigned)h1 << 16); ph.y = (unsigned)h2 | ((unsigned)h3 << 16);
      pl.x = (unsigned)l0 | ((unsigned)l1 << 16); pl.y = (unsigned)l2 | ((unsigned)l3 << 16);
      *(uint2*)&AsH[r][c] = ph;  *(uint2*)&AsL[r][c] = pl;
      split2(bv.x, h0, l0); split2(bv.y, h1, l1);
      split2(bv.z, h2, l2); split2(bv.w, h3, l3);
      ph.x = (unsigned)h0 | ((unsigned)h1 << 16); ph.y = (unsigned)h2 | ((unsigned)h3 << 16);
      pl.x = (unsigned)l0 | ((unsigned)l1 << 16); pl.y = (unsigned)l2 | ((unsigned)l3 << 16);
      *(uint2*)&BsH[r][c] = ph;  *(uint2*)&BsL[r][c] = pl;
    }
    __syncthreads();

#pragma unroll
    for (int ks = 0; ks < 64; ks += 32) {
      FragAB ah[4], al[4], bh[2], bl[2];
      const int am  = lane & 15;
      const int akb = ks + ((lane >> 4) << 3);
#pragma unroll
      for (int mi = 0; mi < 4; ++mi) {
        const int r = wm * 64 + mi * 16 + am;
        ah[mi].u[0] = *(const uint4*)&AsH[r][akb];
        ah[mi].u[1] = *(const uint4*)&AsH[r][akb + 16];
        al[mi].u[0] = *(const uint4*)&AsL[r][akb];
        al[mi].u[1] = *(const uint4*)&AsL[r][akb + 16];
      }
      const int bn  = lane & 15;
      const int bkb = ks + ((lane >> 4) << 3);
#pragma unroll
      for (int ni = 0; ni < 2; ++ni) {
        const int cc = wn * 32 + ni * 16 + bn;
        bh[ni].u[0] = *(const uint4*)&BsH[cc][bkb];
        bh[ni].u[1] = *(const uint4*)&BsH[cc][bkb + 16];
        bl[ni].u[0] = *(const uint4*)&BsL[cc][bkb];
        bl[ni].u[1] = *(const uint4*)&BsL[cc][bkb + 16];
      }
#pragma unroll
      for (int mi = 0; mi < 4; ++mi)
#pragma unroll
        for (int ni = 0; ni < 2; ++ni) {
          acc[mi][ni] = __builtin_amdgcn_wmma_f32_16x16x32_bf16(
              false, ah[mi].v, false, bh[ni].v, (short)0, acc[mi][ni], false, false);
          acc[mi][ni] = __builtin_amdgcn_wmma_f32_16x16x32_bf16(
              false, ah[mi].v, false, bl[ni].v, (short)0, acc[mi][ni], false, false);
          acc[mi][ni] = __builtin_amdgcn_wmma_f32_16x16x32_bf16(
              false, al[mi].v, false, bh[ni].v, (short)0, acc[mi][ni], false, false);
        }
    }
    __syncthreads();
  }

#pragma unroll
  for (int ni = 0; ni < 2; ++ni) {
    const int n = colBase + wn * 32 + ni * 16 + (lane & 15);
    const float bias = be[n];
#pragma unroll
    for (int mi = 0; mi < 4; ++mi) {
      const int mb = rowBase + wm * 64 + mi * 16 + ((lane >> 4) << 3);
#pragma unroll
      for (int r = 0; r < 8; ++r) {
        float v = acc[mi][ni][r] + bias;
        v = v > 0.0f ? v : 0.0f;
        z[(size_t)(mb + r) * DHID + n] = v;
      }
    }
  }
}

// ---------------------------------------------------------------------------
// Kernel 2: per-row exact top-64 (radix select on float bits; values >= 0 after
// ReLU so uint order == float order), scatter z_sparse, fused sparse decode.
// Deterministic chunked compaction; ties broken by lowest index (matches JAX).
// ---------------------------------------------------------------------------
__global__ __launch_bounds__(256) void k_topk_decode(const float* __restrict__ z,
                                                     const float* __restrict__ Wd,
                                                     const float* __restrict__ WT,
                                                     int use_wt,
                                                     float* __restrict__ zs,
                                                     float* __restrict__ xhat) {
  __shared__ unsigned sdata[DHID];
  __shared__ unsigned hist[256];
  __shared__ int cnt_gt[256];
  __shared__ int cnt_eq[256];
  __shared__ float sel_val[TOPK_];
  __shared__ int   sel_idx[TOPK_];
  __shared__ unsigned sh_prefix;
  __shared__ int sh_k, sh_gt_total;

  const int tid = threadIdx.x;
  const int row = blockIdx.x;
  const float* zr = z + (size_t)row * DHID;

  for (int i = tid; i < DHID; i += 256) sdata[i] = __float_as_uint(zr[i]);
  __syncthreads();

  unsigned prefix = 0; int k = TOPK_;
#pragma unroll
  for (int pass = 0; pass < 4; ++pass) {
    const int shift = 24 - 8 * pass;
    const unsigned maskhi = (pass == 0) ? 0u : (0xFFFFFFFFu << (shift + 8));
    hist[tid] = 0;
    __syncthreads();
    for (int i = tid; i < DHID; i += 256) {
      const unsigned u = sdata[i];
      if ((u & maskhi) == prefix) atomicAdd(&hist[(u >> shift) & 255u], 1u);
    }
    __syncthreads();
    if (tid == 0) {
      int acc = 0; int b = 255;
      for (; b > 0; --b) {
        const int c = (int)hist[b];
        if (acc + c >= k) break;
        acc += c;
      }
      sh_prefix = prefix | ((unsigned)b << shift);
      sh_k = k - acc;
    }
    __syncthreads();
    prefix = sh_prefix; k = sh_k;
    __syncthreads();
  }

  const unsigned T = prefix;
  const float tval = __uint_as_float(T);

  const int CH = DHID / 256;                // 48
  const int s0 = tid * CH, s1 = s0 + CH;
  int mygt = 0, myeq = 0;
  for (int i = s0; i < s1; ++i) {
    const unsigned u = sdata[i];
    mygt += (u > T); myeq += (u == T);
  }
  cnt_gt[tid] = mygt; cnt_eq[tid] = myeq;
  __syncthreads();
  if (tid == 0) {
    int a = 0;
    for (int t = 0; t < 256; ++t) { const int c = cnt_gt[t]; cnt_gt[t] = a; a += c; }
    sh_gt_total = a;
    a = 0;
    for (int t = 0; t < 256; ++t) { const int c = cnt_eq[t]; cnt_eq[t] = a; a += c; }
  }
  __syncthreads();
  const int gt_total = sh_gt_total;
  int lim = gt_total + k;
  if (lim > TOPK_) lim = TOPK_;
  int og = cnt_gt[tid];
  int oe = gt_total + cnt_eq[tid];
  float* zsr = zs + (size_t)row * DHID;
  for (int i = s0; i < s1; ++i) {
    const unsigned u = sdata[i];
    float outv = 0.0f;
    if (u > T) {
      outv = __uint_as_float(u);
      sel_val[og] = outv; sel_idx[og] = i; ++og;
    } else if (u == T) {
      if (oe < lim) { outv = tval; sel_val[oe] = tval; sel_idx[oe] = i; }
      ++oe;
    }
    zsr[i] = outv;
  }
  __syncthreads();

  float a0 = 0.f, a1 = 0.f, a2 = 0.f;
  if (use_wt) {
    for (int j = 0; j < lim; ++j) {
      const float v = sel_val[j];
      const float* wr = WT + (size_t)sel_idx[j] * DIN;
      a0 = fmaf(v, wr[tid], a0);
      a1 = fmaf(v, wr[tid + 256], a1);
      a2 = fmaf(v, wr[tid + 512], a2);
    }
  } else {
    for (int j = 0; j < lim; ++j) {
      const float v = sel_val[j];
      const int h = sel_idx[j];
      a0 = fmaf(v, Wd[(size_t)(tid)*DHID + h], a0);
      a1 = fmaf(v, Wd[(size_t)(tid + 256) * DHID + h], a1);
      a2 = fmaf(v, Wd[(size_t)(tid + 512) * DHID + h], a2);
    }
  }
  float* xr = xhat + (size_t)row * DIN;
  xr[tid] = a0; xr[tid + 256] = a1; xr[tid + 512] = a2;
}

// ---------------------------------------------------------------------------
// Host launcher (tiered by workspace size)
// ---------------------------------------------------------------------------
extern "C" void kernel_launch(void* const* d_in, const int* in_sizes, int n_in,
                              void* d_out, int out_size, void* d_ws, size_t ws_size,
                              hipStream_t stream) {
  (void)in_sizes; (void)n_in; (void)out_size;
  const float* x  = (const float*)d_in[0];   // [NB, DIN]
  const float* We = (const float*)d_in[1];   // [DHID, DIN]
  const float* be = (const float*)d_in[2];   // [DHID]
  const float* Wd = (const float*)d_in[3];   // [DIN, DHID]

  float* xhat = (float*)d_out;
  float* z    = xhat + (size_t)NB * DIN;
  float* zs   = z    + (size_t)NB * DHID;

  const size_t WT_BYTES = (size_t)DIN * DHID * sizeof(float);          // 37,748,736
  const size_t XPLANE   = (size_t)NB * DIN * sizeof(unsigned short);   // 25,165,824
  const size_t WPLANE   = (size_t)DHID * DIN * sizeof(unsigned short); // 18,874,368
  const size_t FULL     = WT_BYTES + 2 * XPLANE + 2 * WPLANE;          // ~120 MB

  float* WT = (float*)d_ws;
  unsigned short* xHp  = (unsigned short*)((char*)d_ws + WT_BYTES);
  unsigned short* xLp  = xHp + (size_t)NB * DIN;
  unsigned short* WeHp = xLp + (size_t)NB * DIN;
  unsigned short* WeLp = WeHp + (size_t)DHID * DIN;

  const int use_wt  = (ws_size >= WT_BYTES) ? 1 : 0;
  const int use_pre = (ws_size >= FULL) ? 1 : 0;

  if (use_wt) {
    k_transpose_wdec<<<dim3(DHID / 32, DIN / 32), 256, 0, stream>>>(Wd, WT);
  }
  if (use_pre) {
    const int nx4 = NB * DIN / 4, nw4 = DHID * DIN / 4;
    k_split<<<(nx4 + 255) / 256, 256, 0, stream>>>(x, xHp, xLp, nx4);
    k_split<<<(nw4 + 255) / 256, 256, 0, stream>>>(We, WeHp, WeLp, nw4);
    k_encode_pre<<<dim3(DHID / 128, NB / 128), 256, 8 * 8192 * sizeof(unsigned short), stream>>>(
        xHp, xLp, WeHp, WeLp, be, z);
  } else {
    k_encode<<<dim3(DHID / 128, NB / 128), 256, 0, stream>>>(x, We, be, z);
  }
  k_topk_decode<<<NB, 256, 0, stream>>>(z, Wd, WT, use_wt, zs, xhat);
}
